// HDGN_54546084659307
// MI455X (gfx1250) — matrix operand order repaired and checked by reading, not compiled, not run on hardware
//
#include <hip/hip_runtime.h>
#include <hip/hip_bf16.h>
#include <stdint.h>

// ---------------------------------------------------------------------------
// HDGN (DGCNN-style point net) on MI455X / gfx1250.
// All GEMMs (edge-conv einsums, conv5 embedding, kNN Gram) run on
// v_wmma_f32_16x16x32_bf16 (wave32 WMMA, fp32 accumulate), with 64x64
// register blocking per wave (16 WMMA per 8 fragment loads).
// ---------------------------------------------------------------------------

#define NPTS  1024
#define NB    8
#define NDIRS 8
#define KNN   20

static __device__ __constant__ float kPI    = 3.14159265358979323846f;
#define K_EPS   1.1920928955078125e-07f
#define K_SLOPE 0.2f

typedef __attribute__((ext_vector_type(16))) __bf16 v16bf;
typedef __attribute__((ext_vector_type(8)))  __bf16 v8bf;
typedef __attribute__((ext_vector_type(8)))  float  v8f;

__device__ __forceinline__ v16bf pack16(const __bf16* p0, const __bf16* p1) {
  v8bf lo = *(const v8bf*)p0;
  v8bf hi = *(const v8bf*)p1;
  v16bf r;
#pragma unroll
  for (int i = 0; i < 8; ++i) { r[i] = lo[i]; r[8 + i] = hi[i]; }
  return r;
}

__device__ __forceinline__ v8f wmma_bf16(v16bf a, v16bf b, v8f c) {
  return __builtin_amdgcn_wmma_f32_16x16x32_bf16(false, a, false, b,
                                                 (short)0, c, false, false);
}

// ---------------------------------------------------------------------------
// float (B,C,N) -> bf16 (B,N,Cpad) transpose with zero pad (K-major rows
// so WMMA A/B fragments are contiguous 16-byte loads per lane).
__global__ void transpose_bf16_k(const float* __restrict__ X, __bf16* __restrict__ Xt,
                                 int C, int Cpad) {
  size_t idx = (size_t)blockIdx.x * blockDim.x + threadIdx.x;
  size_t per = (size_t)NPTS * Cpad;
  if (idx >= (size_t)NB * per) return;
  int b = (int)(idx / per);
  size_t r = idx - (size_t)b * per;
  int n = (int)(r / Cpad);
  int c = (int)(r - (size_t)n * Cpad);
  float v = (c < C) ? X[((size_t)b * C + c) * NPTS + n] : 0.0f;
  Xt[idx] = (__bf16)v;
}

// xx[b][n] = sum_c X[b][c][n]^2
__global__ void rowsumsq_k(const float* __restrict__ X, float* __restrict__ xx, int C) {
  int t = blockIdx.x * blockDim.x + threadIdx.x;
  if (t >= NB * NPTS) return;
  int b = t / NPTS, n = t - b * NPTS;
  float s = 0.f;
  for (int c = 0; c < C; ++c) {
    float v = X[((size_t)b * C + c) * NPTS + n];
    s += v * v;
  }
  xx[t] = s;
}

// negd[b][n][m] = 2 * <f_n, f_m> - xx[n] - xx[m]   (Gram via WMMA, 64x64 block)
__global__ void __launch_bounds__(256)
gram_negd_k(const __bf16* __restrict__ Xt, const float* __restrict__ xx,
            float* __restrict__ negd, int Cpad) {
  const int tilesN = NPTS / 64;                  // 16
  const int tilesM = NPTS / 64;                  // 16
  int b    = blockIdx.z;
  int wave = threadIdx.x >> 5;
  int lane = threadIdx.x & 31;
  int tile = blockIdx.x * (blockDim.x >> 5) + wave;
  if (tile >= tilesM * tilesN) return;
  int tm = tile / tilesN, tn = tile - tm * tilesN;
  int n0 = tm * 64, m0 = tn * 64;
  int half = lane >> 4, l16 = lane & 15;

  const __bf16* base = Xt + (size_t)b * NPTS * Cpad;
  const __bf16* arow[4];
#pragma unroll
  for (int mi = 0; mi < 4; ++mi)
    arow[mi] = base + (size_t)(n0 + mi * 16 + l16) * Cpad;

  v8f acc[4][4] = {};
  for (int k0 = 0; k0 < Cpad; k0 += 32) {
    v16bf a[4], bf[4];
#pragma unroll
    for (int mi = 0; mi < 4; ++mi)
      a[mi] = pack16(arow[mi] + k0 + half * 16, arow[mi] + k0 + half * 16 + 8);
#pragma unroll
    for (int j = 0; j < 4; ++j) {
      const __bf16* brow = base + (size_t)(m0 + j * 16 + l16) * Cpad + k0 + half * 16;
      bf[j] = pack16(brow, brow + 8);
    }
#pragma unroll
    for (int mi = 0; mi < 4; ++mi)
#pragma unroll
      for (int j = 0; j < 4; ++j)
        acc[mi][j] = wmma_bf16(a[mi], bf[j], acc[mi][j]);
  }
  const float* xxb = xx + (size_t)b * NPTS;
#pragma unroll
  for (int mi = 0; mi < 4; ++mi)
#pragma unroll
    for (int j = 0; j < 4; ++j) {
      float xm = xxb[m0 + j * 16 + l16];
#pragma unroll
      for (int i = 0; i < 8; ++i) {
        int n = n0 + mi * 16 + i + half * 8;
        negd[((size_t)b * NPTS + n) * NPTS + (m0 + j * 16 + l16)] =
            2.0f * acc[mi][j][i] - xxb[n] - xm;
      }
    }
}

// per-(b,n) top-20 of negd row (descending, ties -> lower index first)
__global__ void topk20_k(const float* __restrict__ negd, int* __restrict__ nn) {
  int t = blockIdx.x * blockDim.x + threadIdx.x;
  if (t >= NB * NPTS) return;
  const float* row = negd + (size_t)t * NPTS;
  float vals[KNN];
  int   idxs[KNN];
#pragma unroll
  for (int k = 0; k < KNN; ++k) { vals[k] = -3.402823466e38f; idxs[k] = 0; }
  for (int m = 0; m < NPTS; ++m) {
    float v = row[m];
    if (v > vals[KNN - 1]) {
      int p = KNN - 1;
      while (p > 0 && vals[p - 1] < v) {
        vals[p] = vals[p - 1]; idxs[p] = idxs[p - 1]; --p;
      }
      vals[p] = v; idxs[p] = m;
    }
  }
  int* out = nn + (size_t)t * KNN;
#pragma unroll
  for (int k = 0; k < KNN; ++k) out[k] = idxs[k];
}

// angle-bin each of 8 standard directions to nearest of the 20 neighbors
__global__ void dirsel_k(const float* __restrict__ xyz, const int* __restrict__ nn,
                         int* __restrict__ dir) {
  int t = blockIdx.x * blockDim.x + threadIdx.x;
  if (t >= NB * NPTS) return;
  int b = t / NPTS, n = t - b * NPTS;
  const float* xb = xyz + (size_t)b * 3 * NPTS;
  float px = xb[n], py = xb[NPTS + n], pz = xb[2 * NPTS + n];
  const int* nnp = nn + (size_t)t * KNN;
  float best[NDIRS];
  int   bi[NDIRS];
#pragma unroll
  for (int j = 0; j < NDIRS; ++j) { best[j] = 3.402823466e38f; bi[j] = 0; }
  for (int k = 0; k < KNN; ++k) {
    int m = nnp[k];
    float vx = px - xb[m];
    float vy = py - xb[NPTS + m];
    float vz = pz - xb[2 * NPTS + m];
    float r  = sqrtf(vx * vx + vy * vy + vz * vz);
    float th = atan2f(vy, vx) + kPI;
    float ph = acosf(vz / (r + K_EPS));
#pragma unroll
    for (int j = 0; j < NDIRS; ++j) {
      float ths = ((float)(j & 3) + 0.5f) * (0.5f * kPI);
      float phs = ((float)(j >> 2) + 0.5f) * (0.5f * kPI);
      float dt = th - ths, dp = ph - phs;
      float d = dt * dt + dp * dp;
      if (d < best[j]) { best[j] = d; bi[j] = k; }
    }
  }
  int* out = dir + (size_t)t * NDIRS;
#pragma unroll
  for (int j = 0; j < NDIRS; ++j) out[j] = nnp[bi[j]];
}

// fp32 weights -> bf16, K padded to Kpad with zeros (row r = blockIdx.y)
__global__ void cvt_w_k(const float* __restrict__ W, __bf16* __restrict__ Wb,
                        int K, int Kpad) {
  int r = blockIdx.y;
  int k = blockIdx.x * blockDim.x + threadIdx.x;
  if (k >= Kpad) return;
  Wb[(size_t)r * Kpad + k] = (k < K) ? (__bf16)W[(size_t)r * K + k] : (__bf16)0.0f;
}

// graph-feature gather, transposed & bf16:
// Gt[j][k], j = n*8 + d, k<C: X[k][m]-X[k][n]; k<2C: X[k-C][n]; else 0
__global__ void gather_gt_k(const float* __restrict__ X, const int* __restrict__ dir,
                            __bf16* __restrict__ Gt, int C, int Kpad, int b) {
  int j = blockIdx.y;
  int k = blockIdx.x * blockDim.x + threadIdx.x;
  if (k >= Kpad) return;
  int n = j >> 3, d = j & 7;
  float v = 0.0f;
  if (k < C) {
    int m = dir[((size_t)b * NPTS + n) * NDIRS + d];
    v = X[(size_t)k * NPTS + m] - X[(size_t)k * NPTS + n];
  } else if (k < 2 * C) {
    v = X[(size_t)(k - C) * NPTS + n];
  }
  Gt[(size_t)j * Kpad + k] = (__bf16)v;
}

// ---------------------------------------------------------------------------
// WMMA GEMM (64x64 per wave): Out(M x Ncols) =
//     lrelu( (A(M x Kpad) * B) * gain[o] + bias[o] )
// bmode 0: B given transposed, bf16 (Ncols x Kpad) row-major (K contiguous)
// bmode 1: B[k=(c*8+d)][n] = H1[c][n*8+d], H1 bf16 (Kpad/8 x Ncols*8)
// out_bf16 1: store bf16 row-major (M x Ncols); else f32.
__global__ void __launch_bounds__(256)
wmma_gemm_k(const __bf16* __restrict__ A, const __bf16* __restrict__ B0,
            const float* __restrict__ gain, const float* __restrict__ bias,
            void* __restrict__ Out, int M, int Ncols, int Kpad,
            int bmode, int out_bf16, long long strideB, long long strideOut) {
  int b    = blockIdx.z;
  int wave = threadIdx.x >> 5;
  int lane = threadIdx.x & 31;
  int tilesM = M >> 6, tilesN = Ncols >> 6;
  int tile = blockIdx.x * (blockDim.x >> 5) + wave;
  if (tile >= tilesM * tilesN) return;
  int tm = tile / tilesN, tn = tile - tm * tilesN;
  int row0 = tm * 64, col0 = tn * 64;
  int half = lane >> 4, l16 = lane & 15;

  const __bf16* Bp = B0 + (size_t)b * strideB;
  const __bf16* arow[4];
#pragma unroll
  for (int mi = 0; mi < 4; ++mi)
    arow[mi] = A + (size_t)(row0 + mi * 16 + l16) * Kpad;

  v8f acc[4][4] = {};
  if (bmode == 0) {
    for (int k0 = 0; k0 < Kpad; k0 += 32) {
      v16bf a[4], bf[4];
#pragma unroll
      for (int mi = 0; mi < 4; ++mi)
        a[mi] = pack16(arow[mi] + k0 + half * 16, arow[mi] + k0 + half * 16 + 8);
#pragma unroll
      for (int j = 0; j < 4; ++j) {
        const __bf16* brow =
            Bp + (size_t)(col0 + j * 16 + l16) * Kpad + k0 + half * 16;
        bf[j] = pack16(brow, brow + 8);
      }
#pragma unroll
      for (int mi = 0; mi < 4; ++mi)
#pragma unroll
        for (int j = 0; j < 4; ++j)
          acc[mi][j] = wmma_bf16(a[mi], bf[j], acc[mi][j]);
    }
  } else {
    int w = Ncols * NDIRS;                     // H1 row width
    for (int k0 = 0; k0 < Kpad; k0 += 32) {
      v16bf a[4], bf[4];
#pragma unroll
      for (int mi = 0; mi < 4; ++mi)
        a[mi] = pack16(arow[mi] + k0 + half * 16, arow[mi] + k0 + half * 16 + 8);
      int c0 = (k0 + half * 16) >> 3;          // even
#pragma unroll
      for (int j = 0; j < 4; ++j) {
        const __bf16* bp =
            Bp + (size_t)c0 * w + (size_t)(col0 + j * 16 + l16) * NDIRS;
        bf[j] = pack16(bp, bp + w);
      }
#pragma unroll
      for (int mi = 0; mi < 4; ++mi)
#pragma unroll
        for (int j = 0; j < 4; ++j)
          acc[mi][j] = wmma_bf16(a[mi], bf[j], acc[mi][j]);
    }
  }

#pragma unroll
  for (int mi = 0; mi < 4; ++mi)
#pragma unroll
    for (int j = 0; j < 4; ++j) {
      int col = col0 + j * 16 + l16;
#pragma unroll
      for (int i = 0; i < 8; ++i) {
        int o = row0 + mi * 16 + i + half * 8;
        float v = acc[mi][j][i] * gain[o] + bias[o];
        v = (v >= 0.0f) ? v : K_SLOPE * v;
        if (out_bf16) {
          ((__bf16*)Out)[(size_t)b * strideOut + (size_t)o * Ncols + col] = (__bf16)v;
        } else {
          ((float*)Out)[(size_t)b * strideOut + (size_t)o * Ncols + col] = v;
        }
      }
    }
}

// channel-block copy for concat: dst[b][coff + c][n] = src[b][c][n]
__global__ void copy_feat_k(const float* __restrict__ src, float* __restrict__ dst,
                            int Csrc, int Cdst, int coff) {
  size_t idx = (size_t)blockIdx.x * blockDim.x + threadIdx.x;
  size_t per = (size_t)Csrc * NPTS;
  if (idx >= (size_t)NB * per) return;
  int b = (int)(idx / per);
  size_t r = idx - (size_t)b * per;
  dst[((size_t)b * Cdst + coff) * NPTS + r] = src[idx];
}

// z[b][o] = max_n H[b][o][n]; z[b][1024+o] = mean_n
__global__ void pool_k(const float* __restrict__ H, float* __restrict__ z) {
  int t = blockIdx.x * blockDim.x + threadIdx.x;
  if (t >= NB * 1024) return;
  int b = t / 1024, o = t - b * 1024;
  const float* row = H + (size_t)t * NPTS;
  float mx = -3.402823466e38f, sm = 0.f;
  for (int n = 0; n < NPTS; ++n) {
    float v = row[n];
    mx = (v > mx) ? v : mx;
    sm += v;
  }
  z[(size_t)b * 2048 + o]        = mx;
  z[(size_t)b * 2048 + 1024 + o] = sm * (1.0f / NPTS);
}

// small dense layer: out[b][o] = f( dot + lin[o] ) etc.
__global__ void dense_k(const float* __restrict__ in, const float* __restrict__ W,
                        const float* __restrict__ lin, const float* __restrict__ gain,
                        const float* __restrict__ bias, float* __restrict__ out,
                        int In, int Outc, int lrelu) {
  int t = blockIdx.x * blockDim.x + threadIdx.x;
  if (t >= NB * Outc) return;
  int b = t / Outc, o = t - b * Outc;
  const float* ip = in + (size_t)b * In;
  const float* wp = W + (size_t)o * In;
  float acc = 0.f;
  for (int k = 0; k < In; ++k) acc += ip[k] * wp[k];
  if (lin)  acc += lin[o];
  if (gain) acc = acc * gain[o] + bias[o];
  if (lrelu) acc = (acc >= 0.f) ? acc : K_SLOPE * acc;
  out[t] = acc;
}

// ---------------------------------------------------------------------------
extern "C" void kernel_launch(void* const* d_in, const int* in_sizes, int n_in,
                              void* d_out, int out_size, void* d_ws, size_t ws_size,
                              hipStream_t stream) {
  (void)in_sizes; (void)n_in; (void)out_size; (void)ws_size;
  const float* x = (const float*)d_in[0];

  // ---- workspace bump allocator -------------------------------------------
  char* wsp = (char*)d_ws;
  auto alloc = [&](size_t bytes) -> void* {
    uintptr_t p = ((uintptr_t)wsp + 255) & ~(uintptr_t)255;
    wsp = (char*)(p + bytes);
    return (void*)p;
  };
  float*  negd = (float*)alloc((size_t)NB * NPTS * NPTS * 4);
  int*    nn   = (int*)  alloc((size_t)NB * NPTS * KNN * 4);
  int*    dir  = (int*)  alloc((size_t)NB * NPTS * NDIRS * 4);
  float*  xx   = (float*)alloc((size_t)NB * NPTS * 4);
  __bf16* Xt   = (__bf16*)alloc((size_t)NB * NPTS * 928 * 2);
  __bf16* Gt   = (__bf16*)alloc((size_t)8192 * 1024 * 2);          // per-batch
  __bf16* H1   = (__bf16*)alloc((size_t)256 * 8192 * 2);           // per-batch
  __bf16* WaBf = (__bf16*)alloc((size_t)1024 * 928 * 2);
  __bf16* WbBf = (__bf16*)alloc((size_t)512 * 2048 * 2);
  float*  x1   = (float*)alloc((size_t)NB * 128 * NPTS * 4);
  float*  x1b  = (float*)alloc((size_t)NB * 128 * NPTS * 4);
  float*  x1c  = (float*)alloc((size_t)NB * 131 * NPTS * 4);
  float*  x2   = (float*)alloc((size_t)NB * 256 * NPTS * 4);
  float*  x2b  = (float*)alloc((size_t)NB * 256 * NPTS * 4);
  float*  x2c  = (float*)alloc((size_t)NB * 387 * NPTS * 4);
  float*  x3   = (float*)alloc((size_t)NB * 512 * NPTS * 4);
  float*  x3b  = (float*)alloc((size_t)NB * 512 * NPTS * 4);
  float*  x3c  = (float*)alloc((size_t)NB * 899 * NPTS * 4);
  float*  z0   = (float*)alloc((size_t)NB * 2048 * 4);
  float*  z1   = (float*)alloc((size_t)NB * 512 * 4);
  float*  z2   = (float*)alloc((size_t)NB * 256 * 4);
  float*  H5   = negd;   // conv5 output reuses negd (same size, negd dead by then)

  // ---- kNN + direction-select stage ---------------------------------------
  auto run_knn = [&](const float* feat, int C, int Cpad) {
    size_t tt = (size_t)NB * NPTS * Cpad;
    transpose_bf16_k<<<(unsigned)((tt + 255) / 256), 256, 0, stream>>>(feat, Xt, C, Cpad);
    rowsumsq_k<<<(NB * NPTS + 255) / 256, 256, 0, stream>>>(feat, xx, C);
    gram_negd_k<<<dim3(32, 1, NB), 256, 0, stream>>>(Xt, xx, negd, Cpad);
    topk20_k<<<(NB * NPTS + 127) / 128, 128, 0, stream>>>(negd, nn);
    dirsel_k<<<(NB * NPTS + 127) / 128, 128, 0, stream>>>(x, nn, dir);
  };

  // ---- edge-conv block (two WMMA GEMMs, per-batch Gt/H1 staging) ----------
  auto run_edge = [&](const float* featIn, int C, int mid, int cout,
                      int iW, float* outFeat) {
    int cin2  = 2 * C;
    int Kpad1 = ((cin2 + 31) / 32) * 32;
    int K2    = mid * NDIRS;
    const float* Wa = (const float*)d_in[iW + 0];
    const float* ga = (const float*)d_in[iW + 1];
    const float* ba = (const float*)d_in[iW + 2];
    const float* Wb = (const float*)d_in[iW + 3];
    const float* gb = (const float*)d_in[iW + 4];
    const float* bb = (const float*)d_in[iW + 5];
    cvt_w_k<<<dim3((Kpad1 + 255) / 256, mid), 256, 0, stream>>>(Wa, WaBf, cin2, Kpad1);
    cvt_w_k<<<dim3((K2 + 255) / 256, cout), 256, 0, stream>>>(Wb, WbBf, K2, K2);
    int tiles1 = (mid / 64) * (8192 / 64);
    int tiles2 = (cout / 64) * (NPTS / 64);
    for (int b = 0; b < NB; ++b) {
      gather_gt_k<<<dim3((Kpad1 + 255) / 256, 8192), 256, 0, stream>>>(
          featIn + (size_t)b * C * NPTS, dir, Gt, C, Kpad1, b);
      wmma_gemm_k<<<dim3((tiles1 + 7) / 8, 1, 1), 256, 0, stream>>>(
          WaBf, Gt, ga, ba, (void*)H1, mid, 8192, Kpad1, /*bmode*/0, /*bf16*/1, 0, 0);
      wmma_gemm_k<<<dim3((tiles2 + 7) / 8, 1, 1), 256, 0, stream>>>(
          WbBf, H1, gb, bb, (void*)(outFeat + (size_t)b * cout * NPTS),
          cout, NPTS, K2, /*bmode*/1, /*bf16*/0, 0, 0);
    }
  };

  auto concat2 = [&](const float* a, int Ca, const float* bsrc, int Cb, float* dst) {
    int Cdst = Ca + Cb;
    size_t ta = (size_t)NB * Ca * NPTS, tb = (size_t)NB * Cb * NPTS;
    copy_feat_k<<<(unsigned)((ta + 255) / 256), 256, 0, stream>>>(a, dst, Ca, Cdst, 0);
    copy_feat_k<<<(unsigned)((tb + 255) / 256), 256, 0, stream>>>(bsrc, dst, Cb, Cdst, Ca);
  };

  // ---- network ------------------------------------------------------------
  run_knn(x, 3, 32);
  run_edge(x,   3,   128, 128, 1, x1);
  run_edge(x1,  128, 128, 128, 7, x1b);
  concat2(x, 3, x1b, 128, x1c);

  run_knn(x1c, 131, 160);
  run_edge(x1c, 131, 128, 256, 13, x2);
  run_edge(x2,  256, 128, 256, 19, x2b);
  concat2(x1c, 131, x2b, 256, x2c);

  run_knn(x2c, 387, 416);
  run_edge(x2c, 387, 256, 512, 25, x3);
  run_edge(x3,  512, 256, 512, 31, x3b);
  concat2(x2c, 387, x3b, 512, x3c);

  // conv5: (B,1024,N) = lrelu(W5 @ x3c * g5 + b5), WMMA over batched Bt=Xt
  {
    const float* W5 = (const float*)d_in[37];
    const float* g5 = (const float*)d_in[38];
    const float* b5 = (const float*)d_in[39];
    size_t tt = (size_t)NB * NPTS * 928;
    transpose_bf16_k<<<(unsigned)((tt + 255) / 256), 256, 0, stream>>>(x3c, Xt, 899, 928);
    cvt_w_k<<<dim3((928 + 255) / 256, 1024), 256, 0, stream>>>(W5, WaBf, 899, 928);
    int tiles = (1024 / 64) * (NPTS / 64);
    wmma_gemm_k<<<dim3((tiles + 7) / 8, 1, NB), 256, 0, stream>>>(
        WaBf, Xt, g5, b5, (void*)H5, 1024, NPTS, 928, /*bmode*/0, /*bf16*/0,
        (long long)NPTS * 928, (long long)1024 * NPTS);
  }

  // pooling + head
  pool_k<<<(NB * 1024 + 127) / 128, 128, 0, stream>>>(H5, z0);
  dense_k<<<(NB * 512 + 127) / 128, 128, 0, stream>>>(
      z0, (const float*)d_in[40], nullptr, (const float*)d_in[41],
      (const float*)d_in[42], z1, 2048, 512, 1);
  dense_k<<<(NB * 256 + 127) / 128, 128, 0, stream>>>(
      z1, (const float*)d_in[43], (const float*)d_in[44], (const float*)d_in[45],
      (const float*)d_in[46], z2, 512, 256, 1);
  dense_k<<<(NB * 40 + 127) / 128, 128, 0, stream>>>(
      z2, (const float*)d_in[47], (const float*)d_in[48], nullptr,
      nullptr, (float*)d_out, 256, 40, 0);
}
